// SCAM_14379550507786
// MI455X (gfx1250) — matrix operand order repaired and checked by reading, not compiled, not run on hardware
//
#include <hip/hip_runtime.h>

// SCAM fused kernel for MI455X (gfx1250, wave32, WMMA).
// One workgroup (8 waves, 256 thr) per (b,h) slice. All intermediates in LDS.
// Matmuls: v_wmma_f32_16x16x32_f16, f32 accumulation. Coalesced b128 HBM I/O.

typedef _Float16 v8h  __attribute__((ext_vector_type(8)));
typedef _Float16 v16h __attribute__((ext_vector_type(16)));
typedef float    v8f  __attribute__((ext_vector_type(8)));

#define PITCH 136                      // f16 row pitch: 272B, 16B-aligned, bank-staggered
#define SCALE 0.08838834764831845f     // 1/sqrt(128)

static __device__ __forceinline__ v16h cat16(v8h a, v8h b) {
  return __builtin_shufflevector(a, b, 0,1,2,3,4,5,6,7,8,9,10,11,12,13,14,15);
}

static __device__ __forceinline__ v8f wmma_(v16h a, v16h b, v8f c) {
  return __builtin_amdgcn_wmma_f32_16x16x32_f16(false, a, false, b, (short)0, c, false, false);
}

// A fragment (16x32 f16): source row-major [M][K]. Lanes 0-15: M=lane, K chunks {kb..kb+7, kb+16..kb+23}.
static __device__ __forceinline__ v16h load_a(const _Float16* src, int pitch, int r0, int k0, int lane) {
  int m = lane & 15, hi = lane >> 4;
  const _Float16* p = src + (r0 + m) * pitch + k0 + hi * 8;
  v8h c0 = *(const v8h*)p;
  v8h c1 = *(const v8h*)(p + 16);
  return cat16(c0, c1);
}

// B fragment (32x16 f16): source stored [N][K] row-major. Lane n holds K chunks {hi*8..hi*8+7, hi*8+16..hi*8+23}.
static __device__ __forceinline__ v16h load_b(const _Float16* src, int pitch, int n0, int k0, int lane) {
  int n = lane & 15, hi = lane >> 4;
  const _Float16* p = src + (n0 + n) * pitch + k0 + hi * 8;
  v8h c0 = *(const v8h*)p;
  v8h c1 = *(const v8h*)(p + 16);
  return cat16(c0, c1);
}

// ---- Phase 1: LayerNorm + f16 cast. 2 threads per row (64 ch each). ----
static __device__ __forceinline__ void ln_cast(const float* xg, const float* sc, const float* bi,
                                               _Float16* lnO, _Float16* xO,
                                               float* red, float* red2, int t) {
  int row = t >> 1, half = t & 1;
  const float4* src = (const float4*)(xg + row * 128 + half * 64);
  float v[64]; float s = 0.f, s2 = 0.f;
#pragma unroll
  for (int i = 0; i < 16; ++i) {
    float4 f = src[i];
    v[i*4+0] = f.x; v[i*4+1] = f.y; v[i*4+2] = f.z; v[i*4+3] = f.w;
    s  += f.x + f.y + f.z + f.w;
    s2 += f.x*f.x + f.y*f.y + f.z*f.z + f.w*f.w;
  }
  red[t] = s; red2[t] = s2;
  __syncthreads();
  float ts  = red[row*2]  + red[row*2+1];
  float ts2 = red2[row*2] + red2[row*2+1];
  float mean = ts * (1.f/128.f);
  float var  = ts2 * (1.f/128.f) - mean * mean;
  float rs = rsqrtf(var + 1e-6f);
  int cb = half * 64;
#pragma unroll
  for (int c = 0; c < 8; ++c) {
    v8h hn, hx;
#pragma unroll
    for (int e = 0; e < 8; ++e) {
      int j = c*8 + e; float x = v[j];
      hx[e] = (_Float16)x;
      hn[e] = (_Float16)((x - mean) * rs * sc[cb + j] + bi[cb + j]);
    }
    *(v8h*)(lnO + row * PITCH + cb + c*8) = hn;
    *(v8h*)(xO  + row * PITCH + cb + c*8) = hx;
  }
}

// ---- Phase 2: projection  D[W,C] = A[W,C] x wT-B + bias.  Per wave: one 16-row band. ----
template <bool TRANS>
static __device__ __forceinline__ void proj_mm(const _Float16* aS, const _Float16* wT, const float* bias,
                                               _Float16* dst, int wave, int lane) {
  int r0 = wave * 16, n = lane & 15, hi = lane >> 4;
  v16h A[4];
#pragma unroll
  for (int k = 0; k < 4; ++k) A[k] = load_a(aS, PITCH, r0, k*32, lane);
#pragma unroll
  for (int tn = 0; tn < 8; ++tn) {
    v8f acc = {};
#pragma unroll
    for (int k = 0; k < 4; ++k) acc = wmma_(A[k], load_b(wT, 128, tn*16, k*32, lane), acc);
    float b = bias[tn*16 + n];
    if (TRANS) {  // store D transposed: contiguous b128 per lane -> vT[c][w]
      v8h h;
#pragma unroll
      for (int g = 0; g < 8; ++g) h[g] = (_Float16)(acc[g] + b);
      *(v8h*)(dst + (tn*16 + n) * PITCH + r0 + hi*8) = h;
    } else {      // store D row-major (column-scatter, conflict-free 2x16 pattern)
#pragma unroll
      for (int g = 0; g < 8; ++g)
        dst[(r0 + g + hi*8) * PITCH + tn*16 + n] = (_Float16)(acc[g] + b);
    }
  }
}

// ---- Phase 3: scores.  D = A x Brows^T, stored TRANSPOSED (contiguous). ----
static __device__ __forceinline__ void score_mm(const _Float16* aS, const _Float16* bS,
                                                _Float16* dT, int wave, int lane) {
  int r0 = wave * 16, n = lane & 15, hi = lane >> 4;
  v16h A[4];
#pragma unroll
  for (int k = 0; k < 4; ++k) A[k] = load_a(aS, PITCH, r0, k*32, lane);
#pragma unroll
  for (int tn = 0; tn < 8; ++tn) {
    v8f acc = {};
#pragma unroll
    for (int k = 0; k < 4; ++k) acc = wmma_(A[k], load_b(bS, PITCH, tn*16, k*32, lane), acc);
    v8h h;
#pragma unroll
    for (int g = 0; g < 8; ++g) h[g] = (_Float16)(acc[g] * SCALE);
    *(v8h*)(dT + (tn*16 + n) * PITCH + r0 + hi*8) = h;
  }
}

// ---- Phase 4: row softmax in place (f16). 2 threads per row. ----
static __device__ __forceinline__ void softmax_rows(_Float16* buf, float* red, float* red2, int t) {
  int row = t >> 1, half = t & 1;
  _Float16* p = buf + row * PITCH + half * 64;
  float v[64]; float m = -1e30f;
#pragma unroll
  for (int c = 0; c < 8; ++c) {
    v8h h = *(const v8h*)(p + c*8);
#pragma unroll
    for (int e = 0; e < 8; ++e) { float f = (float)h[e]; v[c*8+e] = f; m = fmaxf(m, f); }
  }
  red[t] = m;
  __syncthreads();
  m = fmaxf(red[row*2], red[row*2+1]);
  float s = 0.f;
#pragma unroll
  for (int j = 0; j < 64; ++j) { float e = __expf(v[j] - m); v[j] = e; s += e; }
  red2[t] = s;
  __syncthreads();
  float inv = 1.0f / (red2[row*2] + red2[row*2+1]);
#pragma unroll
  for (int c = 0; c < 8; ++c) {
    v8h h;
#pragma unroll
    for (int e = 0; e < 8; ++e) h[e] = (_Float16)(v[c*8+e] * inv);
    *(v8h*)(p + c*8) = h;
  }
}

// ---- Phase 5a: F = P x V, D frags scatter-stored row-major f16 into a dead LDS buffer. ----
static __device__ __forceinline__ void final_store(const _Float16* pS, const _Float16* vT,
                                                   _Float16* fDst, int wave, int lane) {
  int r0 = wave * 16, n = lane & 15, hi = lane >> 4;
  v16h A[4];
#pragma unroll
  for (int k = 0; k < 4; ++k) A[k] = load_a(pS, PITCH, r0, k*32, lane);
#pragma unroll
  for (int tn = 0; tn < 8; ++tn) {
    v8f acc = {};
#pragma unroll
    for (int k = 0; k < 4; ++k) acc = wmma_(A[k], load_b(vT, PITCH, tn*16, k*32, lane), acc);
#pragma unroll
    for (int g = 0; g < 8; ++g)
      fDst[(r0 + g + hi*8) * PITCH + tn*16 + n] = (_Float16)acc[g];
  }
}

// ---- Phase 5b: coalesced epilogue  out = x + F * scaleVec  (b128 global I/O). ----
static __device__ __forceinline__ void epilogue(const _Float16* fS, const float* sc,
                                                const float* xg, float* og, int t) {
  int row = t >> 1, half = t & 1, cb = half * 64;
  const _Float16* fp = fS + row * PITCH + cb;
  const float4* xp = (const float4*)(xg + row * 128 + cb);
  float4* op = (float4*)(og + row * 128 + cb);
#pragma unroll
  for (int c = 0; c < 8; ++c) {
    v8h f = *(const v8h*)(fp + c*8);
    float4 x0 = xp[c*2], x1 = xp[c*2 + 1];
    float4 o0, o1;
    o0.x = x0.x + (float)f[0] * sc[cb + c*8 + 0];
    o0.y = x0.y + (float)f[1] * sc[cb + c*8 + 1];
    o0.z = x0.z + (float)f[2] * sc[cb + c*8 + 2];
    o0.w = x0.w + (float)f[3] * sc[cb + c*8 + 3];
    o1.x = x1.x + (float)f[4] * sc[cb + c*8 + 4];
    o1.y = x1.y + (float)f[5] * sc[cb + c*8 + 5];
    o1.z = x1.z + (float)f[6] * sc[cb + c*8 + 6];
    o1.w = x1.w + (float)f[7] * sc[cb + c*8 + 7];
    op[c*2] = o0; op[c*2 + 1] = o1;
  }
}

// ---- Weight prep: f32 [K][N] -> f16 [N][K] (B-fragment friendly) ----
extern "C" __global__ void __launch_bounds__(256)
scam_prep(const float* __restrict__ wql, const float* __restrict__ wqr,
          const float* __restrict__ wvl, const float* __restrict__ wvr,
          _Float16* __restrict__ wT) {
  int id = blockIdx.x * 256 + threadIdx.x;          // 0..65535
  int mat = id >> 14;
  int ofs = id & 16383;
  int n = ofs >> 7, k = ofs & 127;
  const float* src = (mat == 0) ? wql : (mat == 1) ? wqr : (mat == 2) ? wvl : wvr;
  wT[id] = (_Float16)src[k * 128 + n];
}

extern "C" __global__ void __launch_bounds__(256)
scam_main(const float* __restrict__ x_l, const float* __restrict__ x_r,
          const float* __restrict__ lnls, const float* __restrict__ lnlb,
          const float* __restrict__ lnrs, const float* __restrict__ lnrb,
          const float* __restrict__ bql, const float* __restrict__ bqr,
          const float* __restrict__ bvl, const float* __restrict__ bvr,
          const float* __restrict__ beta, const float* __restrict__ gamma,
          const _Float16* __restrict__ wT,
          float* __restrict__ out_l, float* __restrict__ out_r) {
  __shared__ __align__(16) _Float16 s_ql [128 * PITCH];  // Q_l row-major  -> later F_l
  __shared__ __align__(16) _Float16 s_qr [128 * PITCH];  // Q_r row-major  -> later F_r
  __shared__ __align__(16) _Float16 s_vlT[128 * PITCH];  // V_l transposed [c][w]
  __shared__ __align__(16) _Float16 s_vrT[128 * PITCH];  // V_r transposed [c][w]
  __shared__ __align__(16) _Float16 s_bufA[128 * PITCH]; // ln(x)/x16 -> S    -> P_row
  __shared__ __align__(16) _Float16 s_bufB[128 * PITCH]; // ln(x)/x16 -> S^T  -> P_col^T
  __shared__ float s_par[1280];
  __shared__ float s_red[256], s_red2[256];

  int t = threadIdx.x, lane = t & 31, wave = t >> 5;
  size_t base = (size_t)blockIdx.x * 16384;           // (b,h) slice: 128x128 f32
  const float* xl = x_l + base;
  const float* xr = x_r + base;

  // Warm L2/L0 for the right-side slice while the left side is processed.
  __builtin_prefetch(xr + (t >> 1) * 128 + (t & 1) * 64, 0, 3);

  if (t < 128) {
    s_par[t]        = lnls[t];  s_par[128 + t]  = lnlb[t];
    s_par[256 + t]  = lnrs[t];  s_par[384 + t]  = lnrb[t];
    s_par[512 + t]  = bql[t];   s_par[640 + t]  = bqr[t];
    s_par[768 + t]  = bvl[t];   s_par[896 + t]  = bvr[t];
    s_par[1024 + t] = beta[t];  s_par[1152 + t] = gamma[t];
  }

  // LEFT: LN + cast, then Q_l / V_l projections
  ln_cast(xl, s_par + 0, s_par + 128, s_bufA, s_bufB, s_red, s_red2, t);
  __syncthreads();
  proj_mm<false>(s_bufA, wT,          s_par + 512, s_ql,  wave, lane);
  proj_mm<true >(s_bufB, wT + 32768,  s_par + 768, s_vlT, wave, lane);
  __syncthreads();

  // RIGHT
  ln_cast(xr, s_par + 256, s_par + 384, s_bufA, s_bufB, s_red, s_red2, t);
  __syncthreads();
  proj_mm<false>(s_bufA, wT + 16384,  s_par + 640, s_qr,  wave, lane);
  proj_mm<true >(s_bufB, wT + 49152,  s_par + 896, s_vrT, wave, lane);
  __syncthreads();

  // Scores: Ql.Qr^T stored transposed -> bufB = S^T ; Qr.Ql^T stored transposed -> bufA = S.
  score_mm(s_ql, s_qr, s_bufB, wave, lane);
  score_mm(s_qr, s_ql, s_bufA, wave, lane);
  __syncthreads();

  // Dual softmax = row softmax on S and on S^T, in place (f16).
  softmax_rows(s_bufA, s_red, s_red2, t);
  softmax_rows(s_bufB, s_red, s_red2, t);
  __syncthreads();

  // F_l = P_row x V_r -> s_ql ;  F_r = P_col^T x V_l -> s_qr  (Q buffers are dead now)
  final_store(s_bufA, s_vrT, s_ql, wave, lane);
  final_store(s_bufB, s_vlT, s_qr, wave, lane);
  __syncthreads();

  // Coalesced residual + scale epilogue: out = x + F * {beta,gamma}
  epilogue(s_ql, s_par + 1024, xl, out_l + base, t);
  epilogue(s_qr, s_par + 1152, xr, out_r + base, t);
}

extern "C" void kernel_launch(void* const* d_in, const int* in_sizes, int n_in,
                              void* d_out, int out_size, void* d_ws, size_t ws_size,
                              hipStream_t stream) {
  const float* x_l  = (const float*)d_in[0];
  const float* x_r  = (const float*)d_in[1];
  const float* lnls = (const float*)d_in[2];
  const float* lnlb = (const float*)d_in[3];
  const float* lnrs = (const float*)d_in[4];
  const float* lnrb = (const float*)d_in[5];
  const float* w_ql = (const float*)d_in[6];
  const float* b_ql = (const float*)d_in[7];
  const float* w_qr = (const float*)d_in[8];
  const float* b_qr = (const float*)d_in[9];
  const float* w_vl = (const float*)d_in[10];
  const float* b_vl = (const float*)d_in[11];
  const float* w_vr = (const float*)d_in[12];
  const float* b_vr = (const float*)d_in[13];
  const float* beta = (const float*)d_in[14];
  const float* gmma = (const float*)d_in[15];

  _Float16* wT = (_Float16*)d_ws;                    // 4 x 128x128 f16 = 128 KB
  float* out_l = (float*)d_out;
  float* out_r = out_l + (size_t)8 * 128 * 128 * 128;

  scam_prep<<<256, 256, 0, stream>>>(w_ql, w_qr, w_vl, w_vr, wT);
  scam_main<<<8 * 128, 256, 0, stream>>>(x_l, x_r, lnls, lnlb, lnrs, lnrb,
                                         b_ql, b_qr, b_vl, b_vr, beta, gmma,
                                         wT, out_l, out_r);
}
